// MultiHeadAttention_17042430231324
// MI455X (gfx1250) — matrix operand order repaired and checked by reading, not compiled, not run on hardware
//
#include <hip/hip_runtime.h>
#include <hip/hip_bf16.h>

typedef _Float16 f16;
typedef __attribute__((ext_vector_type(16))) _Float16 v16h;
typedef __attribute__((ext_vector_type(8)))  _Float16 v8h;
typedef __attribute__((ext_vector_type(8)))  float    v8f;

// D = A(16x32 f16) * B(32x16 f16) + C(16x16 f32), wave32 WMMA
__device__ __forceinline__ v8f wmma_f16(v16h a, v16h b, v8f c) {
  return __builtin_amdgcn_wmma_f32_16x16x32_f16(false, a, false, b, (short)0, c,
                                                false, false);
}

// Generic pointer (known to point into LDS) -> LDS-relative byte offset,
// as required by the VDST operand of global_load_async_to_lds_*.
__device__ __forceinline__ uint32_t lds_addr(const void* p) {
  return (uint32_t)(uintptr_t)(__attribute__((address_space(3))) const void*)p;
}

// Async global->LDS copy of 16B per lane (GLOBAL_LOAD_ASYNC_TO_LDS_B128,
// tracked by ASYNCcnt). Portable across toolchains via inline asm.
__device__ __forceinline__ void async_g2l_b128(uint32_t lds_off,
                                               const void* gaddr) {
  asm volatile("global_load_async_to_lds_b128 %0, %1, off"
               :: "v"(lds_off), "v"(gaddr)
               : "memory");
}

// A-fragment (16x32, 16-bit): lane holds row = lane%16.
// element j <-> K = (lane/16)*16 + j  => two contiguous 16B loads.
__device__ __forceinline__ v16h loadA16(const f16* __restrict__ base, int ld,
                                        int row, int k0, int hi) {
  const f16* p = base + (size_t)row * ld + k0 + hi * 16;
  v8h lo = *(const v8h*)(p);
  v8h hh = *(const v8h*)(p + 8);
  v16h a;
#pragma unroll
  for (int j = 0; j < 8; ++j) { a[j] = lo[j]; a[j + 8] = hh[j]; }
  return a;
}

// ---------------- conversion kernels ----------------
__global__ void f32_to_f16_kernel(const float* __restrict__ in,
                                  f16* __restrict__ out, int n) {
  int i = blockIdx.x * blockDim.x + threadIdx.x;
  if (i < n) out[i] = (f16)in[i];
}

// Wt[n][k] = W[k][n], 1024x1024, fp32 -> f16
__global__ void transpose_w_kernel(const float* __restrict__ in,
                                   f16* __restrict__ out) {
  int i = blockIdx.x * blockDim.x + threadIdx.x;  // 0..1M-1
  int n = i >> 10, k = i & 1023;
  out[i] = (f16)in[k * 1024 + n];
}

// ---------------- WMMA GEMM: C[M=4096][N=1024] = A*Wt^T + bias --------------
// B tile (64 cols x 32 k) double-buffered in LDS via async global->LDS loads;
// shared by all 8 waves of the block (kills the 8x redundant global traffic).
// MODE 0: f16 out scattered to Q/K layout [B][H][S][64]
// MODE 1: f16 out scattered to V^T layout [B][H][64][S]
// MODE 2: f32 out row-major [4096][1024] (final projection)
template <int MODE>
__global__ __launch_bounds__(256) void gemm_wmma_kernel(
    const f16* __restrict__ A, const f16* __restrict__ Bt,
    const float* __restrict__ bias, void* __restrict__ outp) {
  __shared__ __align__(64) f16 ldsB[2][64][32];  // 8 KB, double buffered
  const int lane = threadIdx.x & 31;
  const int wave = threadIdx.x >> 5;
  const int l16 = lane & 15;
  const int hi = lane >> 4;
  const int m0 = blockIdx.x * 128 + wave * 16;
  const int n0 = blockIdx.y * 64;
  const int sn = threadIdx.x >> 2;        // staging row (output column) 0..63
  const int sc = (threadIdx.x & 3) * 8;   // staging f16 offset within row

  // prologue: stage buffer 0 (k0 = 0)
  async_g2l_b128(lds_addr(&ldsB[0][sn][sc]),
                 Bt + (size_t)(n0 + sn) * 1024 + sc);

  v8f c[4] = {};
  for (int k0 = 0; k0 < 1024; k0 += 32) {
    const int cur = (k0 >> 5) & 1;
    if (k0 + 32 < 1024) {
      // stage next tile, then wait for the *current* one (in-order returns:
      // <=1 outstanding async op per wave => older stage complete).
      async_g2l_b128(lds_addr(&ldsB[cur ^ 1][sn][sc]),
                     Bt + (size_t)(n0 + sn) * 1024 + (k0 + 32) + sc);
      asm volatile("s_wait_asynccnt 1" ::: "memory");
      __builtin_prefetch(A + (size_t)(m0 + l16) * 1024 + k0 + 32, 0, 3);
    } else {
      asm volatile("s_wait_asynccnt 0" ::: "memory");
    }
    __syncthreads();

    v16h a = loadA16(A, 1024, m0 + l16, k0, hi);
#pragma unroll
    for (int t = 0; t < 4; ++t) {
      // B-frag: lane col = t*16+l16, elements K = hi*16 + j, contiguous in LDS
      v16h b = *(const v16h*)&ldsB[cur][t * 16 + l16][hi * 16];
      c[t] = wmma_f16(a, b, c[t]);
    }
    __syncthreads();  // safe to overwrite ldsB[cur] next iteration
  }

#pragma unroll
  for (int t = 0; t < 4; ++t) {
    const int gc = n0 + t * 16 + l16;
    const float bb = bias[gc];
#pragma unroll
    for (int r = 0; r < 8; ++r) {
      const int gr = m0 + hi * 8 + r;  // D element r -> row r + 8*(lane/16)
      const float val = c[t][r] + bb;
      if (MODE == 2) {
        ((float*)outp)[(size_t)gr * 1024 + gc] = val;
      } else {
        const int b = gr >> 11, s = gr & 2047;
        const int h = gc >> 6, d = gc & 63;
        size_t addr;
        if (MODE == 0)
          addr = (((size_t)(b * 16 + h) * 2048) + s) * 64 + d;
        else
          addr = (((size_t)(b * 16 + h) * 64) + d) * 2048 + s;
        ((f16*)outp)[addr] = (f16)val;
      }
    }
  }
}

// ---------------- flash attention --------------------------------------
// grid = (S/64, B*H), block = 128 (4 waves). Wave w owns 16 query rows,
// full d_k=64 output. K/V 64-wide tiles staged to LDS with async loads and
// shared by all 4 waves; online softmax over 64-key blocks (causal).
__global__ __launch_bounds__(128) void flash_attn_kernel(
    const f16* __restrict__ Q, const f16* __restrict__ K,
    const f16* __restrict__ Vt, const float* __restrict__ pad,
    f16* __restrict__ O) {
  __shared__ __align__(64) f16 ldsP[4][16][64];  // per-wave P tile (8 KB)
  __shared__ __align__(64) f16 ldsK[64][64];     // K tile  [key][d]   (8 KB)
  __shared__ __align__(64) f16 ldsV[64][64];     // V^T tile [d][key]  (8 KB)
  const int lane = threadIdx.x & 31;
  const int wave = threadIdx.x >> 5;
  const int l16 = lane & 15;
  const int hi = lane >> 4;
  const int qblk = blockIdx.x;
  const int bh = blockIdx.y;
  const int b = bh >> 4;
  const int h = bh & 15;
  const f16* Qh = Q + (size_t)bh * 2048 * 64;
  const f16* Kh = K + (size_t)bh * 2048 * 64;
  const f16* Vh = Vt + (size_t)bh * 64 * 2048;
  const int q0 = qblk * 64 + wave * 16;

  v16h aq0 = loadA16(Qh, 64, q0 + l16, 0, hi);
  v16h aq1 = loadA16(Qh, 64, q0 + l16, 32, hi);

  float mrow[8], lrow[8];
  v8f acc[4] = {};
#pragma unroll
  for (int r = 0; r < 8; ++r) { mrow[r] = -3.0e38f; lrow[r] = 0.0f; }

  for (int kb = 0; kb <= qblk; ++kb) {
    const int kbase = kb * 64;
    // ---- async-stage K and V^T tiles (512 x 16B chunks each -> 4/thread) --
#pragma unroll
    for (int u = 0; u < 4; ++u) {
      const int chunk = u * 128 + threadIdx.x;  // 0..511
      const int row = chunk >> 3;
      const int off = (chunk & 7) * 8;
      async_g2l_b128(lds_addr(&ldsK[row][off]),
                     Kh + (size_t)(kbase + row) * 64 + off);
      async_g2l_b128(lds_addr(&ldsV[row][off]),
                     Vh + (size_t)row * 2048 + kbase + off);
    }
    asm volatile("s_wait_asynccnt 0" ::: "memory");
    __syncthreads();

    float p[4][8];
    // ---- scores S = (Q K^T) * 1/sqrt(dk) + masks ----
#pragma unroll
    for (int t = 0; t < 4; ++t) {
      const int key = kbase + t * 16 + l16;
      v16h b0 = *(const v16h*)&ldsK[t * 16 + l16][hi * 16];       // d  0..31
      v16h b1 = *(const v16h*)&ldsK[t * 16 + l16][32 + hi * 16];  // d 32..63
      v8f c = {};
      c = wmma_f16(aq0, b0, c);
      c = wmma_f16(aq1, b1, c);
      const float pm = pad[b * 2048 + key];
#pragma unroll
      for (int r = 0; r < 8; ++r) {
        float s = c[r] * 0.125f + pm;  // 1/sqrt(64)
        if (kb == qblk && key > q0 + hi * 8 + r) s = -1.0e30f;  // causal
        p[t][r] = s;
      }
    }
    // ---- online softmax update (row spans the 16 lanes of one half) ----
#pragma unroll
    for (int r = 0; r < 8; ++r) {
      float tm = fmaxf(fmaxf(p[0][r], p[1][r]), fmaxf(p[2][r], p[3][r]));
#pragma unroll
      for (int o = 1; o < 16; o <<= 1) tm = fmaxf(tm, __shfl_xor(tm, o, 32));
      const float mn = fmaxf(mrow[r], tm);
      const float corr = __expf(mrow[r] - mn);
      float rs = 0.0f;
#pragma unroll
      for (int t = 0; t < 4; ++t) {
        const float pe = __expf(p[t][r] - mn);
        p[t][r] = pe;
        rs += pe;
      }
#pragma unroll
      for (int o = 1; o < 16; o <<= 1) rs += __shfl_xor(rs, o, 32);
      lrow[r] = lrow[r] * corr + rs;
      mrow[r] = mn;
#pragma unroll
      for (int t = 0; t < 4; ++t) acc[t][r] *= corr;
    }
    // ---- reshape P: C-layout regs -> LDS -> A-layout frags ----
#pragma unroll
    for (int t = 0; t < 4; ++t)
#pragma unroll
      for (int r = 0; r < 8; ++r)
        ldsP[wave][hi * 8 + r][t * 16 + l16] = (f16)p[t][r];
    __syncthreads();
    v16h pa0, pa1;
#pragma unroll
    for (int j = 0; j < 8; ++j) {
      pa0[j]     = ldsP[wave][l16][hi * 16 + j];
      pa0[j + 8] = ldsP[wave][l16][hi * 16 + 8 + j];
      pa1[j]     = ldsP[wave][l16][32 + hi * 16 + j];
      pa1[j + 8] = ldsP[wave][l16][32 + hi * 16 + 8 + j];
    }
    // ---- O += P * V (V^T tile in LDS => contiguous B-frags) ----
#pragma unroll
    for (int t = 0; t < 4; ++t) {
      v16h bv0 = *(const v16h*)&ldsV[t * 16 + l16][hi * 16];       // keys  0..31
      v16h bv1 = *(const v16h*)&ldsV[t * 16 + l16][32 + hi * 16];  // keys 32..63
      acc[t] = wmma_f16(pa0, bv0, acc[t]);
      acc[t] = wmma_f16(pa1, bv1, acc[t]);
    }
    __syncthreads();  // protect ldsK/ldsV/ldsP before next iteration
  }

  // ---- normalize, scatter heads back to [B*S][D] f16 ----
#pragma unroll
  for (int r = 0; r < 8; ++r) {
    const float inv = 1.0f / lrow[r];
    const int s = q0 + hi * 8 + r;
#pragma unroll
    for (int t = 0; t < 4; ++t) {
      O[(size_t)(b * 2048 + s) * 1024 + h * 64 + t * 16 + l16] =
          (f16)(acc[t][r] * inv);
    }
  }
}

extern "C" void kernel_launch(void* const* d_in, const int* in_sizes, int n_in,
                              void* d_out, int out_size, void* d_ws,
                              size_t ws_size, hipStream_t stream) {
  (void)in_sizes; (void)n_in; (void)out_size; (void)ws_size;
  const float* q32 = (const float*)d_in[0];
  const float* k32 = (const float*)d_in[1];
  const float* v32 = (const float*)d_in[2];
  const float* pad = (const float*)d_in[3];
  // d_in[4] lookahead mask: causal mask applied analytically
  const float* wq = (const float*)d_in[5];
  const float* wk = (const float*)d_in[6];
  const float* wv = (const float*)d_in[7];
  const float* wo = (const float*)d_in[8];
  const float* bq = (const float*)d_in[9];
  const float* bk = (const float*)d_in[10];
  const float* bv = (const float*)d_in[11];
  const float* bo = (const float*)d_in[12];

  const int NTOK = 4096;               // B*S
  const int NELT = NTOK * 1024;        // 4,194,304
  const int WELT = 1024 * 1024;

  f16* qf  = (f16*)d_ws;               // activations f16
  f16* kf  = qf + NELT;
  f16* vf  = kf + NELT;
  f16* wqT = vf + NELT;                // weights transposed f16 [N][K]
  f16* wkT = wqT + WELT;
  f16* wvT = wkT + WELT;
  f16* woT = wvT + WELT;
  f16* Qh  = woT + WELT;               // [B][H][S][64]
  f16* Kh  = Qh + NELT;                // [B][H][S][64]
  f16* Vth = Kh + NELT;                // [B][H][64][S]
  f16* Ob  = Vth + NELT;               // [B*S][D]

  f32_to_f16_kernel<<<NELT / 256, 256, 0, stream>>>(q32, qf, NELT);
  f32_to_f16_kernel<<<NELT / 256, 256, 0, stream>>>(k32, kf, NELT);
  f32_to_f16_kernel<<<NELT / 256, 256, 0, stream>>>(v32, vf, NELT);
  transpose_w_kernel<<<WELT / 256, 256, 0, stream>>>(wq, wqT);
  transpose_w_kernel<<<WELT / 256, 256, 0, stream>>>(wk, wkT);
  transpose_w_kernel<<<WELT / 256, 256, 0, stream>>>(wv, wvT);
  transpose_w_kernel<<<WELT / 256, 256, 0, stream>>>(wo, woT);

  dim3 ggrid(NTOK / 128, 1024 / 64);   // (32, 16)
  gemm_wmma_kernel<0><<<ggrid, 256, 0, stream>>>(qf, wqT, bq, (void*)Qh);
  gemm_wmma_kernel<0><<<ggrid, 256, 0, stream>>>(kf, wkT, bk, (void*)Kh);
  gemm_wmma_kernel<1><<<ggrid, 256, 0, stream>>>(vf, wvT, bv, (void*)Vth);

  dim3 agrid(2048 / 64, 32);           // (q-blocks, B*H)
  flash_attn_kernel<<<agrid, 128, 0, stream>>>(Qh, Kh, Vth, pad, Ob);

  gemm_wmma_kernel<2><<<ggrid, 256, 0, stream>>>(Ob, woT, bo, d_out);
}
